// GraphAttentionBlock_13709535609673
// MI455X (gfx1250) — matrix operand order repaired and checked by reading, not compiled, not run on hardware
//
#include <hip/hip_runtime.h>
#include <hip/hip_bf16.h>
#include <math.h>

typedef __attribute__((ext_vector_type(16))) _Float16 v16h;
typedef __attribute__((ext_vector_type(8)))  _Float16 v8h;
typedef __attribute__((ext_vector_type(8)))  float    v8f;

#define BB 4
#define NN 2048
#define DIMD 512
#define NHEAD 8
#define HD 64
#define MROWS (BB * NN)   /* 8192 */
#define FFN 2048

__device__ __forceinline__ v8f wmma_f16(v16h a, v16h b, v8f c) {
  return __builtin_amdgcn_wmma_f32_16x16x32_f16(false, a, false, b, (short)0, c, false, false);
}

// A-fragment (16x32 f16, MxK): lane = M row (lane&15); lane half selects K 0-7 vs 8-15
// in VGPR0-3 and K 16-23 vs 24-31 in VGPR4-7. Two B128 loads per lane.
__device__ __forceinline__ v16h load_a_frag(const _Float16* p, int lda) {
  const int lane = threadIdx.x & 31;
  const int m = lane & 15, hh = lane >> 4;
  const _Float16* row = p + (size_t)m * lda + hh * 8;
  v8h lo = *(const v8h*)(row);
  v8h hi = *(const v8h*)(row + 16);
  v16h r;
#pragma unroll
  for (int e = 0; e < 8; ++e) { r[e] = lo[e]; r[e + 8] = hi[e]; }
  return r;
}

// B-fragment (32x16 f16, KxN) read from B^T (N x K row-major): lane = N col (lane&15);
// lane half selects K 0-7 vs 8-15 in VGPR0-3 and K 16-23 vs 24-31 in VGPR4-7; two B128 loads.
__device__ __forceinline__ v16h load_b_frag(const _Float16* p, int ldb) {
  const int lane = threadIdx.x & 31;
  const int n = lane & 15, hh = lane >> 4;
  const _Float16* row = p + (size_t)n * ldb + hh * 8;
  v8h lo = *(const v8h*)(row);
  v8h hi = *(const v8h*)(row + 16);
  v16h r;
#pragma unroll
  for (int e = 0; e < 8; ++e) { r[e] = lo[e]; r[e + 8] = hi[e]; }
  return r;
}

// ---------- weight transpose + f16 cast: in (K x Nc) f32 -> out (Nc x K) f16 ----------
__global__ void wcast_kernel(const float* __restrict__ in, _Float16* __restrict__ out,
                             int K, int Nc) {
  size_t idx = (size_t)blockIdx.x * blockDim.x + threadIdx.x;
  if (idx >= (size_t)K * Nc) return;
  int k = (int)(idx / Nc), n = (int)(idx % Nc);
  out[(size_t)n * K + k] = (_Float16)in[idx];
}

// ---------- layernorm over last dim 512 + f16 cast ----------
__global__ void __launch_bounds__(256) ln_cast_kernel(const float* __restrict__ in,
    const float* __restrict__ g, const float* __restrict__ bta,
    _Float16* __restrict__ out) {
  __shared__ float red[256];
  const int t = threadIdx.x;
  const size_t row = blockIdx.x;
  const float* x = in + row * DIMD;
  float x0 = x[t], x1 = x[t + 256];
  red[t] = x0 + x1;
  __syncthreads();
  for (int o = 128; o > 0; o >>= 1) { if (t < o) red[t] += red[t + o]; __syncthreads(); }
  float mu = red[0] * (1.0f / DIMD);
  __syncthreads();
  float d0 = x0 - mu, d1 = x1 - mu;
  red[t] = d0 * d0 + d1 * d1;
  __syncthreads();
  for (int o = 128; o > 0; o >>= 1) { if (t < o) red[t] += red[t + o]; __syncthreads(); }
  float rs = rsqrtf(red[0] * (1.0f / DIMD) + 1e-5f);
  _Float16* o16 = out + row * DIMD;
  o16[t]       = (_Float16)(d0 * rs * g[t] + bta[t]);
  o16[t + 256] = (_Float16)(d1 * rs * g[t + 256] + bta[t + 256]);
}

// ---------- generic WMMA GEMM ----------
// out(M x Nc) = A(M x K, f16) @ Bt(Nc x K, f16)^T + bias
// Block: 128 threads = 4 waves, block tile 64(M) x 128(N); wave tile 32 x 64 (2x4 WMMA subtiles).
// epi: 0 = store f16 ; 1 = GELU then store f16 ; 2 = fp32 store with residual add ;
//      3 = QKV scatter to Q(b,h,N,d) / K(b,h,N,d) / V^T(b,h,d,N) f16
__global__ void __launch_bounds__(128) gemm_kernel(const _Float16* __restrict__ A,
    const _Float16* __restrict__ Bt, const float* __restrict__ bias,
    void* __restrict__ outp, const float* __restrict__ resid,
    _Float16* __restrict__ Qs, _Float16* __restrict__ Ks, _Float16* __restrict__ Vts,
    int M, int Nc, int K, int epi) {
  const int wave = threadIdx.x >> 5;
  const int row0 = blockIdx.y * 64 + (wave >> 1) * 32;
  const int col0 = blockIdx.x * 128 + (wave & 1) * 64;
  v8f acc[2][4] = {};
  for (int k0 = 0; k0 < K; k0 += 32) {
    v16h a0 = load_a_frag(A + (size_t)row0 * K + k0, K);
    v16h a1 = load_a_frag(A + (size_t)(row0 + 16) * K + k0, K);
    v16h b0 = load_b_frag(Bt + (size_t)(col0 +  0) * K + k0, K);
    v16h b1 = load_b_frag(Bt + (size_t)(col0 + 16) * K + k0, K);
    v16h b2 = load_b_frag(Bt + (size_t)(col0 + 32) * K + k0, K);
    v16h b3 = load_b_frag(Bt + (size_t)(col0 + 48) * K + k0, K);
    acc[0][0] = wmma_f16(a0, b0, acc[0][0]);
    acc[0][1] = wmma_f16(a0, b1, acc[0][1]);
    acc[0][2] = wmma_f16(a0, b2, acc[0][2]);
    acc[0][3] = wmma_f16(a0, b3, acc[0][3]);
    acc[1][0] = wmma_f16(a1, b0, acc[1][0]);
    acc[1][1] = wmma_f16(a1, b1, acc[1][1]);
    acc[1][2] = wmma_f16(a1, b2, acc[1][2]);
    acc[1][3] = wmma_f16(a1, b3, acc[1][3]);
  }
  const int lane = threadIdx.x & 31;
  const int n = lane & 15, hh = lane >> 4;
#pragma unroll
  for (int i = 0; i < 2; ++i) {
#pragma unroll
    for (int j = 0; j < 4; ++j) {
      int c = col0 + j * 16 + n;
      float bv = bias[c];
#pragma unroll
      for (int r = 0; r < 8; ++r) {
        int rr = row0 + i * 16 + hh * 8 + r;
        float v = acc[i][j][r] + bv;
        if (epi == 0) {
          ((_Float16*)outp)[(size_t)rr * Nc + c] = (_Float16)v;
        } else if (epi == 1) {
          ((_Float16*)outp)[(size_t)rr * Nc + c] =
              (_Float16)(0.5f * v * (1.0f + erff(v * 0.70710678118f)));
        } else if (epi == 2) {
          size_t idx = (size_t)rr * Nc + c;
          ((float*)outp)[idx] = resid[idx] + v;
        } else {
          // QKV scatter: rr = b*N + i_tok ; c in [0, 1536)
          int b  = rr >> 11, itok = rr & (NN - 1);
          int which = c / DIMD, rem = c % DIMD;
          int h = rem >> 6, d = rem & 63;
          size_t bh = (size_t)(b * NHEAD + h);
          _Float16 hv = (_Float16)v;
          if (which == 0)      Qs[(bh * NN + itok) * HD + d] = hv;
          else if (which == 1) Ks[(bh * NN + itok) * HD + d] = hv;
          else                 Vts[(bh * HD + d) * NN + itok] = hv;
        }
      }
    }
  }
}

// ---------- flash attention with edge bias ----------
// grid: (B*H, N/64), block 128 (4 waves; each wave owns 16 query rows)
__global__ void __launch_bounds__(128) attn_kernel(const _Float16* __restrict__ Q,
    const _Float16* __restrict__ Kd, const _Float16* __restrict__ Vt,
    const float* __restrict__ Ae, const float* __restrict__ edge_w,
    const float* __restrict__ edge_scale, _Float16* __restrict__ outp) {
  __shared__ _Float16 plds[4][16 * 32];
  const int wave = threadIdx.x >> 5, lane = threadIdx.x & 31;
  const int n = lane & 15, hh = lane >> 4;
  const int bh = blockIdx.x, b = bh >> 3, h = bh & 7;
  const int i0 = blockIdx.y * 64 + wave * 16;
  const float cb = edge_scale[0] * edge_w[h];
  const float scale = 0.125f;  // 64^-0.5
  const _Float16* Qb = Q + ((size_t)bh * NN + i0) * HD;
  const _Float16* Kb = Kd + (size_t)bh * NN * HD;
  const _Float16* Vb = Vt + (size_t)bh * HD * NN;
  const float* Ab = Ae + (size_t)b * NN * NN;
  v16h aq0 = load_a_frag(Qb, HD);
  v16h aq1 = load_a_frag(Qb + 32, HD);
  float mrow[8], lrow[8];
#pragma unroll
  for (int r = 0; r < 8; ++r) { mrow[r] = -3.0e38f; lrow[r] = 0.0f; }
  v8f acc[4] = {};
  _Float16* my = plds[wave];
  for (int j0 = 0; j0 < NN; j0 += 32) {
    v8f S[2] = {};
#pragma unroll
    for (int s = 0; s < 2; ++s) {
      S[s] = wmma_f16(aq0, load_b_frag(Kb + (size_t)(j0 + s * 16) * HD, HD), S[s]);
      S[s] = wmma_f16(aq1, load_b_frag(Kb + (size_t)(j0 + s * 16) * HD + 32, HD), S[s]);
    }
    float tmax[8];
#pragma unroll
    for (int r = 0; r < 8; ++r) tmax[r] = -3.0e38f;
#pragma unroll
    for (int s = 0; s < 2; ++s) {
#pragma unroll
      for (int r = 0; r < 8; ++r) {
        int i = i0 + hh * 8 + r;
        int j = j0 + s * 16 + n;
        float v = S[s][r] * scale + cb * Ab[(size_t)i * NN + j];
        S[s][r] = v;
        tmax[r] = fmaxf(tmax[r], v);
      }
    }
#pragma unroll
    for (int r = 0; r < 8; ++r) {
#pragma unroll
      for (int mk = 1; mk < 16; mk <<= 1)
        tmax[r] = fmaxf(tmax[r], __shfl_xor(tmax[r], mk, 32));
      float mnew = fmaxf(mrow[r], tmax[r]);
      float alpha = __expf(mrow[r] - mnew);
      mrow[r] = mnew;
      lrow[r] *= alpha;
#pragma unroll
      for (int t = 0; t < 4; ++t) acc[t][r] *= alpha;
    }
    float psum[8];
#pragma unroll
    for (int r = 0; r < 8; ++r) psum[r] = 0.0f;
#pragma unroll
    for (int s = 0; s < 2; ++s) {
#pragma unroll
      for (int r = 0; r < 8; ++r) {
        float p = __expf(S[s][r] - mrow[r]);
        psum[r] += p;
        my[(hh * 8 + r) * 32 + s * 16 + n] = (_Float16)p;
      }
    }
#pragma unroll
    for (int r = 0; r < 8; ++r) {
#pragma unroll
      for (int mk = 1; mk < 16; mk <<= 1)
        psum[r] += __shfl_xor(psum[r], mk, 32);
      lrow[r] += psum[r];
    }
    asm volatile("s_wait_dscnt 0" ::: "memory");
    v16h ap = load_a_frag(my, 32);
#pragma unroll
    for (int t = 0; t < 4; ++t)
      acc[t] = wmma_f16(ap, load_b_frag(Vb + (size_t)(t * 16) * NN + j0, NN), acc[t]);
  }
#pragma unroll
  for (int t = 0; t < 4; ++t) {
#pragma unroll
    for (int r = 0; r < 8; ++r) {
      int i = i0 + hh * 8 + r;
      int d = t * 16 + n;
      outp[((size_t)b * NN + i) * DIMD + h * HD + d] = (_Float16)(acc[t][r] / lrow[r]);
    }
  }
}

extern "C" void kernel_launch(void* const* d_in, const int* in_sizes, int n_in,
                              void* d_out, int out_size, void* d_ws, size_t ws_size,
                              hipStream_t stream) {
  const float* Hin    = (const float*)d_in[0];
  const float* Ae     = (const float*)d_in[1];
  const float* ln1_g  = (const float*)d_in[2];
  const float* ln1_b  = (const float*)d_in[3];
  const float* qkv_w  = (const float*)d_in[4];
  const float* qkv_b  = (const float*)d_in[5];
  const float* proj_w = (const float*)d_in[6];
  const float* proj_b = (const float*)d_in[7];
  const float* ln2_g  = (const float*)d_in[8];
  const float* ln2_b  = (const float*)d_in[9];
  const float* w1     = (const float*)d_in[10];
  const float* b1     = (const float*)d_in[11];
  const float* w2     = (const float*)d_in[12];
  const float* b2     = (const float*)d_in[13];
  const float* edge_w = (const float*)d_in[14];
  const float* edge_s = (const float*)d_in[15];
  float* out = (float*)d_out;
  (void)in_sizes; (void)n_in; (void)out_size; (void)ws_size;

  char* ws = (char*)d_ws;
  size_t off = 0;
  auto alloc = [&](size_t bytes) -> void* {
    void* p = (void*)(ws + off);
    off += (bytes + 255) & ~(size_t)255;
    return p;
  };
  _Float16* Xf16    = (_Float16*)alloc((size_t)MROWS * DIMD * 2);
  _Float16* qkv_wt  = (_Float16*)alloc((size_t)3 * DIMD * DIMD * 2);
  _Float16* proj_wt = (_Float16*)alloc((size_t)DIMD * DIMD * 2);
  _Float16* w1t     = (_Float16*)alloc((size_t)FFN * DIMD * 2);
  _Float16* w2t     = (_Float16*)alloc((size_t)DIMD * FFN * 2);
  _Float16* Qh      = (_Float16*)alloc((size_t)MROWS * DIMD * 2);
  _Float16* Kh      = (_Float16*)alloc((size_t)MROWS * DIMD * 2);
  _Float16* Vt      = (_Float16*)alloc((size_t)MROWS * DIMD * 2);
  _Float16* attn_o  = (_Float16*)alloc((size_t)MROWS * DIMD * 2);
  float*    H1      = (float*)alloc((size_t)MROWS * DIMD * 4);
  _Float16* X2      = (_Float16*)alloc((size_t)MROWS * DIMD * 2);
  _Float16* ffnmid  = (_Float16*)alloc((size_t)MROWS * FFN * 2);

  // 1) transpose+cast all weights to f16 (N x K row-major)
  { int tot = DIMD * 3 * DIMD;
    wcast_kernel<<<(tot + 255) / 256, 256, 0, stream>>>(qkv_w, qkv_wt, DIMD, 3 * DIMD); }
  { int tot = DIMD * DIMD;
    wcast_kernel<<<(tot + 255) / 256, 256, 0, stream>>>(proj_w, proj_wt, DIMD, DIMD); }
  { int tot = DIMD * FFN;
    wcast_kernel<<<(tot + 255) / 256, 256, 0, stream>>>(w1, w1t, DIMD, FFN); }
  { int tot = FFN * DIMD;
    wcast_kernel<<<(tot + 255) / 256, 256, 0, stream>>>(w2, w2t, FFN, DIMD); }

  // 2) LN1 -> f16
  ln_cast_kernel<<<MROWS, 256, 0, stream>>>(Hin, ln1_g, ln1_b, Xf16);

  // 3) QKV GEMM (8192 x 1536, K=512), epilogue scatters straight to Q / K / V^T
  gemm_kernel<<<dim3(3 * DIMD / 128, MROWS / 64), 128, 0, stream>>>(
      Xf16, qkv_wt, qkv_b, nullptr, nullptr, Qh, Kh, Vt, MROWS, 3 * DIMD, DIMD, 3);

  // 4) flash attention with edge bias
  attn_kernel<<<dim3(BB * NHEAD, NN / 64), 128, 0, stream>>>(
      Qh, Kh, Vt, Ae, edge_w, edge_s, attn_o);

  // 5) proj GEMM + residual (fp32)
  gemm_kernel<<<dim3(DIMD / 128, MROWS / 64), 128, 0, stream>>>(
      attn_o, proj_wt, proj_b, H1, Hin, nullptr, nullptr, nullptr, MROWS, DIMD, DIMD, 2);

  // 6) LN2 -> f16
  ln_cast_kernel<<<MROWS, 256, 0, stream>>>(H1, ln2_g, ln2_b, X2);

  // 7) FFN up GEMM + exact GELU -> f16
  gemm_kernel<<<dim3(FFN / 128, MROWS / 64), 128, 0, stream>>>(
      X2, w1t, b1, ffnmid, nullptr, nullptr, nullptr, nullptr, MROWS, FFN, DIMD, 1);

  // 8) FFN down GEMM + residual -> fp32 output
  gemm_kernel<<<dim3(DIMD / 128, MROWS / 64), 128, 0, stream>>>(
      ffnmid, w2t, b2, out, H1, nullptr, nullptr, nullptr, MROWS, DIMD, FFN, 2);
}
